// ScatterAttention_19920058319221
// MI455X (gfx1250) — matrix operand order repaired and checked by reading, not compiled, not run on hardware
//
#include <hip/hip_runtime.h>
#include <hip/hip_bf16.h>
#include <math.h>

// Problem constants (from the reference)
static constexpr int Bc  = 4;
static constexpr int Nc  = 200000;
static constexpr int Ec  = 128;
static constexpr int IDc = 2048;

typedef __attribute__((ext_vector_type(16))) __bf16 v16bf;
typedef __attribute__((ext_vector_type(4)))  __bf16 v4bf;
typedef __attribute__((ext_vector_type(8)))  float  v8f;

// Hardware tanh (V_TANH_F32 transcendental on gfx1250), with safe fallbacks.
__device__ __forceinline__ float hw_tanh(float x) {
#if __has_builtin(__builtin_amdgcn_tanhf)
    return __builtin_amdgcn_tanhf(x);
#elif __has_builtin(__builtin_amdgcn_tanh_f32)
    return __builtin_amdgcn_tanh_f32(x);
#else
    return tanhf(x);
#endif
}

// ---- monotone float<->uint mapping for atomicMax on floats -----------------
__device__ __forceinline__ unsigned enc_f32(float f) {
    unsigned u = __float_as_uint(f);
    return (u & 0x80000000u) ? ~u : (u | 0x80000000u);
}
__device__ __forceinline__ float dec_f32(unsigned u) {
    unsigned v = (u & 0x80000000u) ? (u & 0x7FFFFFFFu) : ~u;
    return __uint_as_float(v);
}

// ---------------------------------------------------------------------------
// K1: importance[b,n] = tanh(x·W^T + bias)·context / sqrt(E), masked.
// 8 waves per block, one 16-row tile per wave, bf16 WMMA with f32 accumulate.
// ---------------------------------------------------------------------------
__global__ __launch_bounds__(256)
void importance_kernel(const float* __restrict__ x, const float* __restrict__ W,
                       const float* __restrict__ bias, const float* __restrict__ ctx,
                       const int* __restrict__ mask, float* __restrict__ score)
{
    __shared__ __bf16 Wl[Ec * Ec];     // W row-major [f][e], bf16 (32 KB)
    __shared__ float  bl[Ec], cl[Ec];

    const int tid = threadIdx.x;
    // Vectorized W staging: float4 in, packed 4x bf16 (b64) out. 16 iterations.
    for (int i = tid * 4; i < Ec * Ec; i += 256 * 4) {
        const float4 w4 = *(const float4*)(W + i);
        v4bf p;
        p[0] = (__bf16)w4.x; p[1] = (__bf16)w4.y;
        p[2] = (__bf16)w4.z; p[3] = (__bf16)w4.w;
        *(v4bf*)(&Wl[i]) = p;
    }
    if (tid < Ec) { bl[tid] = bias[tid]; cl[tid] = ctx[tid]; }
    __syncthreads();

    const int b    = blockIdx.y;
    const int wave = tid >> 5;
    const int lane = tid & 31;
    const int n0   = blockIdx.x * 128 + wave * 16;   // tile row base (wave-uniform)
    if (n0 >= Nc) return;                            // wave-uniform exit: EXEC stays full

    const int M = lane & 15;      // A row / B column / D column selector
    const int h = lane >> 4;      // half-wave

    // Row r of D depends only on row r of A, so OOB rows may hold garbage:
    // clamp the address (always-valid unconditional loads), guard only the store.
    const int row  = n0 + M;
    const int rowc = row < Nc ? row : (Nc - 1);
    const float* xrow = x + ((size_t)b * Nc + rowc) * Ec;

    // A fragments: ISA 16-bit A 16x32 layout collapses to two contiguous
    // 8-float chunks per K-step: [kk+8h, +8) and [kk+16+8h, +8).
    v16bf afrag[4];
    #pragma unroll
    for (int s = 0; s < 4; ++s) {
        const int kk = 32 * s;
        const float4 c0 = *(const float4*)(xrow + kk + 8 * h);
        const float4 c1 = *(const float4*)(xrow + kk + 8 * h + 4);
        const float4 c2 = *(const float4*)(xrow + kk + 16 + 8 * h);
        const float4 c3 = *(const float4*)(xrow + kk + 16 + 8 * h + 4);
        const float cf[16] = {c0.x, c0.y, c0.z, c0.w, c1.x, c1.y, c1.z, c1.w,
                              c2.x, c2.y, c2.z, c2.w, c3.x, c3.y, c3.z, c3.w};
        #pragma unroll
        for (int j = 0; j < 16; ++j) afrag[s][j] = (__bf16)cf[j];
    }

    v8f acc[8];
    #pragma unroll
    for (int t = 0; t < 8; ++t) acc[t] = (v8f){0,0,0,0,0,0,0,0};

    #pragma unroll
    for (int s = 0; s < 4; ++s) {
        const int kk = 32 * s;
        #pragma unroll
        for (int t = 0; t < 8; ++t) {
            // B fragment: B[k][f] = W[f][k]; lane holds column f = t*16 + (lane&15),
            // k chunks [8h, +8) and [16+8h, +8) per half-wave (matches A).
            v16bf bb;
            const __bf16* src = &Wl[(t * 16 + M) * Ec + kk];
            #pragma unroll
            for (int j = 0; j < 16; ++j) bb[j] = src[j < 8 ? 8 * h + j : 16 + 8 * h + (j - 8)];
            acc[t] = __builtin_amdgcn_wmma_f32_16x16x32_bf16(
                         false, afrag[s], false, bb, (short)0, acc[t], false, false);
        }
    }

    // D layout: VGPR r -> row M=r+8h, col f = t*16 + (lane&15).
    const int fl = lane & 15;

    // Hoist this lane's bias/context values (fixed column set across all rows).
    float blr[8], clr[8];
    #pragma unroll
    for (int t = 0; t < 8; ++t) {
        blr[t] = bl[t * 16 + fl];
        clr[t] = cl[t * 16 + fl];
    }

    float sred[8];
    #pragma unroll
    for (int r = 0; r < 8; ++r) {
        float acc_r = 0.0f;
        #pragma unroll
        for (int t = 0; t < 8; ++t)
            acc_r += hw_tanh(acc[t][r] + blr[t]) * clr[t];
        sred[r] = acc_r;
    }
    // reduce across the 16 lanes of each half-wave
    #pragma unroll
    for (int r = 0; r < 8; ++r) {
        #pragma unroll
        for (int off = 1; off < 16; off <<= 1)
            sred[r] += __shfl_xor(sred[r], off, 32);
    }
    if (fl == 0) {
        const float inv_norm = 0.08838834764831845f;   // 1/sqrt(128)
        #pragma unroll
        for (int r = 0; r < 8; ++r) {
            const int n = n0 + 8 * h + r;
            if (n < Nc) {
                float v = sred[r] * inv_norm;
                if (mask[(size_t)b * Nc + n] == 0) v = -1e9f;
                score[(size_t)b * Nc + n] = v;
            }
        }
    }
}

// K2: zero accumulation targets (must run every call — scatter accumulates).
__global__ void zero_kernel(float* __restrict__ out,
                            unsigned* __restrict__ segmax,
                            float* __restrict__ segsum)
{
    const int i = blockIdx.x * blockDim.x + threadIdx.x;
    if (i < Bc * IDc * Ec) out[i] = 0.0f;
    if (i < Bc * IDc) { segmax[i] = 0u; segsum[i] = 0.0f; }
}

// K3: per-segment max via monotone-uint atomicMax.
__global__ void segmax_kernel(const float* __restrict__ score,
                              const int* __restrict__ index,
                              unsigned* __restrict__ segmax)
{
    const size_t i = (size_t)blockIdx.x * blockDim.x + threadIdx.x;
    if (i >= (size_t)Bc * Nc) return;
    const int b   = (int)(i / Nc);
    const int seg = index[i];
    atomicMax(&segmax[b * IDc + seg], enc_f32(score[i]));
}

// K4: e = exp(s - m) in place; segment sums via atomicAdd (global_atomic_add_f32).
__global__ void expsum_kernel(float* __restrict__ score,
                              const int* __restrict__ index,
                              const unsigned* __restrict__ segmax,
                              float* __restrict__ segsum)
{
    const size_t i = (size_t)blockIdx.x * blockDim.x + threadIdx.x;
    if (i >= (size_t)Bc * Nc) return;
    const int b   = (int)(i / Nc);
    const int seg = index[i];
    float m = dec_f32(segmax[b * IDc + seg]);
    // NaN (empty segment, init value) fails both compares -> m = 0, matching isfinite guard
    if (!(m > -3.0e38f && m < 3.0e38f)) m = 0.0f;
    const float e = expf(score[i] - m);
    score[i] = e;
    atomicAdd(&segsum[b * IDc + seg], e);
}

// K5: one block per row: w = e/sum, scatter w*x into out[seg], write attn = w.
__global__ __launch_bounds__(128)
void scatter_kernel(const float* __restrict__ x, const int* __restrict__ index,
                    const float* __restrict__ segsum,
                    float* __restrict__ out, float* __restrict__ attn)
{
    const size_t rowId = blockIdx.x;              // over B*N
    const int b   = (int)(rowId / Nc);
    const int seg = index[rowId];
    const float e = attn[rowId];                  // attn buffer currently holds e
    const float w = e / segsum[b * IDc + seg];
    __syncthreads();                              // all reads of e before the in-place write
    const int f = threadIdx.x;
    atomicAdd(&out[((size_t)(b * IDc + seg)) * Ec + f], x[rowId * Ec + f] * w);
    if (f == 0) attn[rowId] = w;
}

extern "C" void kernel_launch(void* const* d_in, const int* in_sizes, int n_in,
                              void* d_out, int out_size, void* d_ws, size_t ws_size,
                              hipStream_t stream)
{
    const float* x    = (const float*)d_in[0];
    const float* W    = (const float*)d_in[1];
    const float* bias = (const float*)d_in[2];
    const float* ctx  = (const float*)d_in[3];
    const int*   mask = (const int*)d_in[4];
    const int*   idx  = (const int*)d_in[5];

    float* out  = (float*)d_out;                    // [B, IDX, E]
    float* attn = out + (size_t)Bc * IDc * Ec;      // [B, N] — also score/e staging

    unsigned* segmax = (unsigned*)d_ws;             // [B*IDX]
    float*    segsum = (float*)(segmax + Bc * IDc); // [B*IDX]

    // K1: WMMA importance -> attn buffer (scores)
    dim3 g1((Nc + 127) / 128, Bc);
    importance_kernel<<<g1, 256, 0, stream>>>(x, W, bias, ctx, mask, attn);

    // K2: zero out + segment state
    zero_kernel<<<(Bc * IDc * Ec + 255) / 256, 256, 0, stream>>>(out, segmax, segsum);

    // K3/K4: segment max, then exp + segment sum
    const int totalBN = Bc * Nc;
    segmax_kernel<<<(totalBN + 255) / 256, 256, 0, stream>>>(attn, idx, segmax);
    expsum_kernel<<<(totalBN + 255) / 256, 256, 0, stream>>>(attn, idx, segmax, segsum);

    // K5: weighted scatter + final attention weights
    scatter_kernel<<<totalBN, 128, 0, stream>>>(x, idx, segsum, out, attn);
}
